// STTNSNet_22058952032496
// MI455X (gfx1250) — compile-verified, matrix operand order checked by reading, not run on hardware
//
#include <hip/hip_runtime.h>
#include <math.h>

// ---------------------------------------------------------------------------
// STTNSNet forward for MI455X (gfx1250, wave32, WMMA).
// All matmuls use v_wmma_f32_16x16x32_bf16 (fp32 accum, bf16 operands).
// A and B operands are staged in LDS so that each WMMA fragment half is a
// contiguous 16-byte run -> ds_load_b128 (B is stored transposed in LDS).
// Attention score matrices live entirely in LDS (no 144MB S round-trip).
// ---------------------------------------------------------------------------

typedef __attribute__((ext_vector_type(16))) __bf16 v16bf;
typedef __attribute__((ext_vector_type(8)))  float  v8f;

#define NB   8      // B
#define NN   150    // N
#define NT   100    // T
#define NE   64     // E
#define NH   2      // H
#define NPD  32     // PD
#define KTOP 20
#define BNT  (NB*NN*NT)       // 120000
#define BNTH (BNT*NH)         // 240000

// ======================= generic batched WMMA GEMM =========================
// C[z] = act( A[z] (MxK, lda) @ B[z] (KxN or NxK if transB, ldb) + bias )
// BM=32 x BN=64 tile per block; 8 waves, one 16x16 WMMA subtile each.
// Blds is stored TRANSPOSED ([col][k]) so fragment loads are contiguous.
#define GBM 32
#define GBN 64
#define GKMAX 256

__global__ __launch_bounds__(256) void gemm_wmma_kernel(
    const float* __restrict__ A, const float* __restrict__ B, float* __restrict__ C,
    const float* __restrict__ bias,
    int M, int Ncols, int K, int lda, int ldb, int ldc,
    long long sA, long long sB, long long sC,
    int transB, int act, int bias_mask)   // bias index = gcol & bias_mask
{
    __shared__ alignas(16) __bf16 Alds[GBM * GKMAX];   // [row][k]
    __shared__ alignas(16) __bf16 Blds[GBN * GKMAX];   // [col][k]  (transposed)

    const int z = blockIdx.z;
    const float* Ab = A + (long long)z * sA;
    const float* Bb = B + (long long)z * sB;
    float*       Cb = C + (long long)z * sC;

    const int row0 = blockIdx.x * GBM;
    const int col0 = blockIdx.y * GBN;
    const int tid  = threadIdx.x;
    const int KP   = (K + 31) & ~31;

    // stage A [GBM][KP]
    for (int r = 0; r < GBM; ++r) {
        int gr = row0 + r;
        for (int c = tid; c < KP; c += 256) {
            float v = 0.0f;
            if (gr < M && c < K) v = Ab[(long long)gr * lda + c];
            Alds[r * KP + c] = (__bf16)v;
        }
    }
    // stage B transposed: Blds[c][k]
    {
        const int cB = tid & 63;
        const int gc = col0 + cB;
        for (int k = tid >> 6; k < KP; k += 4) {
            float v = 0.0f;
            if (k < K && gc < Ncols)
                v = transB ? Bb[(long long)gc * ldb + k]
                           : Bb[(long long)k  * ldb + gc];
            Blds[cB * KP + k] = (__bf16)v;
        }
    }
    __syncthreads();

    const int w    = tid >> 5;
    const int lane = tid & 31;
    const int mi   = (w & 1) * 16;
    const int ni   = (w >> 1) * 16;
    const int l15  = lane & 15;
    const int kb8  = (lane >> 4) * 8;

    v8f acc = {0.f,0.f,0.f,0.f,0.f,0.f,0.f,0.f};
    const __bf16* aprow = Alds + (mi + l15) * KP;
    const __bf16* bprow = Blds + (ni + l15) * KP;

    for (int kt = 0; kt < KP; kt += 32) {
        v16bf af, bfv;
        const __bf16* ap = aprow + kt + kb8;
        const __bf16* bp = bprow + kt + kb8;
#pragma unroll
        for (int e = 0; e < 8; ++e) { af[e] = ap[e];      bfv[e] = bp[e]; }
#pragma unroll
        for (int e = 0; e < 8; ++e) { af[8+e] = ap[16+e]; bfv[8+e] = bp[16+e]; }
        acc = __builtin_amdgcn_wmma_f32_16x16x32_bf16(
                  false, af, false, bfv, (short)0, acc, false, false);
    }

    const int gcol = col0 + ni + l15;
    if (gcol < Ncols) {
        float bv = bias ? bias[gcol & bias_mask] : 0.0f;
#pragma unroll
        for (int r = 0; r < 8; ++r) {
            int grow = row0 + mi + r + ((lane >> 4) << 3);
            if (grow < M) {
                float v = acc[r] + bv;
                if (act == 1)      v = v > 0.0f ? v : 0.0f;     // relu
                else if (act == 2) v = tanhf(3.0f * v);         // tanh(alpha*x)
                Cb[(long long)grow * ldc + gcol] = v;
            }
        }
    }
}

static void gemm(hipStream_t st, const float* A, const float* B, float* C,
                 const float* bias, int M, int Ncols, int K,
                 int lda, int ldb, int ldc,
                 int batch, long long sA, long long sB, long long sC,
                 int transB, int act, int bias_mod /* 0 or power of 2 */)
{
    int mask = bias_mod ? (bias_mod - 1) : 0x7fffffff;
    dim3 g((M + GBM - 1) / GBM, (Ncols + GBN - 1) / GBN, batch);
    gemm_wmma_kernel<<<g, dim3(256), 0, st>>>(A, B, C, bias, M, Ncols, K,
                                              lda, ldb, ldc, sA, sB, sC,
                                              transB, act, mask);
}

// ====================== fused attention (S stays in LDS) ===================
// One block per (batch-head). S = Q K^T (WMMA) -> column softmax over the
// query axis (reference softmaxes axis=q) -> O = att V (WMMA).
// RP = rows padded to multiple of 16 (160 spatial, 112 temporal).
// V is stored TRANSPOSED in LDS ([d][row]) for contiguous fragment loads.
template <int RP>
__global__ __launch_bounds__(256) void attn_fused_kernel(
    const float* __restrict__ Q, const float* __restrict__ Km,
    const float* __restrict__ V, float* __restrict__ O,
    int rows, int d1, int d2,
    long long s0, long long s1, long long s2,
    int rstride, float scale)
{
    __shared__ alignas(16) float  S[RP * RP];
    __shared__ alignas(16) __bf16 Qb[RP * NPD];    // [row][d]
    __shared__ alignas(16) __bf16 Kb[RP * NPD];    // [row][d]
    __shared__ alignas(16) __bf16 Vt[NPD * RP];    // [d][row]  (transposed)

    const int z  = blockIdx.x;
    const int i0 = z / d1, rem = z % d1, i1 = rem / d2, i2 = rem % d2;
    const long long base = (long long)i0 * s0 + (long long)i1 * s1 + (long long)i2 * s2;
    const int tid = threadIdx.x;

    for (int i = tid; i < RP * NPD; i += 256) {
        int r = i >> 5, c = i & 31;
        float q = 0.f, k = 0.f, v = 0.f;
        if (r < rows) {
            long long off = base + (long long)r * rstride + c;
            q = Q[off]; k = Km[off]; v = V[off];
        }
        Qb[i] = (__bf16)q;
        Kb[i] = (__bf16)k;
        Vt[c * RP + r] = (__bf16)v;
    }
    __syncthreads();

    const int w = tid >> 5, lane = tid & 31;
    const int l15 = lane & 15, kb8 = (lane >> 4) * 8;
    const int NTL = RP / 16;

    // ---- S = Q @ K^T (d = 32, one WMMA per 16x16 tile) ----
    for (int tile = w; tile < NTL * NTL; tile += 8) {
        const int mt = (tile / NTL) * 16, nt = (tile % NTL) * 16;
        v8f acc = {0.f,0.f,0.f,0.f,0.f,0.f,0.f,0.f};
        v16bf af, bfv;
        const __bf16* ap = Qb + (mt + l15) * NPD + kb8;
        const __bf16* bp = Kb + (nt + l15) * NPD + kb8;   // B[k][n] = K[n][k]
#pragma unroll
        for (int e = 0; e < 8; ++e) { af[e] = ap[e];      bfv[e] = bp[e]; }
#pragma unroll
        for (int e = 0; e < 8; ++e) { af[8+e] = ap[16+e]; bfv[8+e] = bp[16+e]; }
        acc = __builtin_amdgcn_wmma_f32_16x16x32_bf16(
                  false, af, false, bfv, (short)0, acc, false, false);
#pragma unroll
        for (int r = 0; r < 8; ++r)
            S[(mt + r + ((lane >> 4) << 3)) * RP + nt + l15] = acc[r];
    }
    __syncthreads();

    // ---- softmax over query axis (normalize each column over rows) ----
    for (int col = tid; col < rows; col += 256) {
        float m = -3.4e38f;
        for (int q = 0; q < rows; ++q) {
            float v = S[q * RP + col] * scale;
            m = v > m ? v : m;
        }
        float s = 0.f;
        for (int q = 0; q < rows; ++q) {
            float e = __expf(S[q * RP + col] * scale - m);
            S[q * RP + col] = e;
            s += e;
        }
        float inv = 1.0f / s;
        for (int q = 0; q < rows; ++q) S[q * RP + col] *= inv;
    }
    __syncthreads();

    // ---- O = att @ V  (K-dim = RP, V padded rows are zero) ----
    for (int tile = w; tile < NTL * 2; tile += 8) {
        const int mt = (tile >> 1) * 16, nt = (tile & 1) * 16;
        v8f acc = {0.f,0.f,0.f,0.f,0.f,0.f,0.f,0.f};
        const float*  aprow = S  + (mt + l15) * RP;
        const __bf16* bprow = Vt + (nt + l15) * RP;
        for (int kt = 0; kt < RP; kt += 32) {
            v16bf af, bfv;
            const float*  ap = aprow + kt + kb8;
            const __bf16* bp = bprow + kt + kb8;
#pragma unroll
            for (int e = 0; e < 8; ++e) { af[e] = (__bf16)ap[e];      bfv[e] = bp[e]; }
#pragma unroll
            for (int e = 0; e < 8; ++e) { af[8+e] = (__bf16)ap[16+e]; bfv[8+e] = bp[16+e]; }
            acc = __builtin_amdgcn_wmma_f32_16x16x32_bf16(
                      false, af, false, bfv, (short)0, acc, false, false);
        }
        const int gcol = nt + l15;
#pragma unroll
        for (int r = 0; r < 8; ++r) {
            int grow = mt + r + ((lane >> 4) << 3);
            if (grow < rows)
                O[base + (long long)grow * rstride + gcol] = acc[r];
        }
    }
}

// ============================ small kernels ================================
__device__ __forceinline__ float wave_sum(float v) {
#pragma unroll
    for (int o = 16; o; o >>= 1) v += __shfl_xor(v, o, 32);
    return v;
}
__device__ __forceinline__ float wave_max(float v) {
#pragma unroll
    for (int o = 16; o; o >>= 1) v = fmaxf(v, __shfl_xor(v, o, 32));
    return v;
}

__global__ __launch_bounds__(256) void conv1_kernel(
    const float* __restrict__ x, const float* __restrict__ w,
    const float* __restrict__ b, float* __restrict__ q, int n)
{
    int i = blockIdx.x * 256 + threadIdx.x;
    if (i >= n) return;
    int e = i & 63, bnt = i >> 6;
    q[i] = x[bnt] * w[e] + b[e];
}

__global__ __launch_bounds__(256) void a_build_kernel(
    const float* __restrict__ M1, float* __restrict__ a)
{
    int i = blockIdx.x * 256 + threadIdx.x;
    if (i >= NN * NN) return;
    int r = i / NN, c = i % NN;
    float v = tanhf(3.0f * (M1[r * NN + c] - M1[c * NN + r]));
    a[i] = v > 0.0f ? v : 0.0f;
}

__global__ void topk_kernel(const float* __restrict__ a,
                            float* __restrict__ adj, float* __restrict__ work)
{
    int i = blockIdx.x * blockDim.x + threadIdx.x;
    if (i >= NN) return;
    const float* ar = a + i * NN;
    float* wr = work + i * NN;
    float* jr = adj + i * NN;
    for (int j = 0; j < NN; ++j) { wr[j] = ar[j]; jr[j] = 0.0f; }
    for (int p = 0; p < KTOP; ++p) {
        int am = 0; float mv = wr[0];
        for (int j = 1; j < NN; ++j) if (wr[j] > mv) { mv = wr[j]; am = j; }
        jr[am] = ar[am];
        wr[am] = -1e30f;
    }
}

__global__ void stats_kernel(const float* __restrict__ adj, float* __restrict__ st)
{
    __shared__ float ss[256], s2[256];
    float a = 0.f, b = 0.f;
    for (int i = threadIdx.x; i < NN * NN; i += 256) {
        float v = adj[i]; a += v; b += v * v;
    }
    ss[threadIdx.x] = a; s2[threadIdx.x] = b;
    __syncthreads();
    for (int o = 128; o; o >>= 1) {
        if (threadIdx.x < o) { ss[threadIdx.x] += ss[threadIdx.x + o];
                               s2[threadIdx.x] += s2[threadIdx.x + o]; }
        __syncthreads();
    }
    if (threadIdx.x == 0) {
        float m = ss[0] / (float)(NN * NN);
        float var = s2[0] / (float)(NN * NN) - m * m;
        st[0] = m; st[1] = rsqrtf(var + 1e-5f);
    }
}

__global__ __launch_bounds__(256) void adjnorm_kernel(
    const float* __restrict__ adj, const float* __restrict__ st,
    float* __restrict__ adjn)
{
    int i = blockIdx.x * 256 + threadIdx.x;
    if (i >= NN * NN) return;
    adjn[i] = (adj[i] - st[0]) * st[1];
}

// out = x + rowvec  (sel=0: D_S indexed by n; sel=1: temb indexed by t)
__global__ __launch_bounds__(256) void add_bcast_kernel(
    const float* __restrict__ x, const float* __restrict__ tab,
    float* __restrict__ out, int n, int sel)
{
    int i = blockIdx.x * 256 + threadIdx.x;
    if (i >= n) return;
    int e = i & 63, nt = i >> 6;
    int idx = sel ? (nt % NT) : ((nt / NT) % NN);
    out[i] = x[i] + tab[idx * 64 + e];
}

// LayerNorm over E=64 of (a + b + c + d), wave per row; b/c/d optional.
__global__ __launch_bounds__(256) void ln_kernel(
    const float* __restrict__ a, const float* __restrict__ b,
    const float* __restrict__ c, const float* __restrict__ d,
    const float* __restrict__ g, const float* __restrict__ bt,
    float* __restrict__ out, int nrows)
{
    int row = blockIdx.x * 8 + (threadIdx.x >> 5);
    if (row >= nrows) return;
    int lane = threadIdx.x & 31;
    long long base = (long long)row * 64;
    float v0 = a[base + lane], v1 = a[base + lane + 32];
    if (b) { v0 += b[base + lane]; v1 += b[base + lane + 32]; }
    if (c) { v0 += c[base + lane]; v1 += c[base + lane + 32]; }
    if (d) { v0 += d[base + lane]; v1 += d[base + lane + 32]; }
    float s  = wave_sum(v0 + v1);
    float s2 = wave_sum(v0 * v0 + v1 * v1);
    float m   = s * (1.0f / 64.0f);
    float var = s2 * (1.0f / 64.0f) - m * m;
    float inv = rsqrtf(var + 1e-5f);
    out[base + lane]      = (v0 - m) * inv * g[lane]      + bt[lane];
    out[base + lane + 32] = (v1 - m) * inv * g[lane + 32] + bt[lane + 32];
}

// out1 = LN( sigmoid(g1+g2)*Us + (1-sigmoid)*XG + q ; bn1 )
__global__ __launch_bounds__(256) void gate_ln_kernel(
    const float* __restrict__ g1, const float* __restrict__ g2,
    const float* __restrict__ Us, const float* __restrict__ XG,
    const float* __restrict__ q,
    const float* __restrict__ g, const float* __restrict__ bt,
    float* __restrict__ out, int nrows)
{
    int row = blockIdx.x * 8 + (threadIdx.x >> 5);
    if (row >= nrows) return;
    int lane = threadIdx.x & 31;
    long long base = (long long)row * 64;
    float gv0 = 1.0f / (1.0f + __expf(-(g1[base + lane]      + g2[base + lane])));
    float gv1 = 1.0f / (1.0f + __expf(-(g1[base + lane + 32] + g2[base + lane + 32])));
    float v0 = gv0 * Us[base + lane]      + (1.0f - gv0) * XG[base + lane]      + q[base + lane];
    float v1 = gv1 * Us[base + lane + 32] + (1.0f - gv1) * XG[base + lane + 32] + q[base + lane + 32];
    float s  = wave_sum(v0 + v1);
    float s2 = wave_sum(v0 * v0 + v1 * v1);
    float m   = s * (1.0f / 64.0f);
    float var = s2 * (1.0f / 64.0f) - m * m;
    float inv = rsqrtf(var + 1e-5f);
    out[base + lane]      = (v0 - m) * inv * g[lane]      + bt[lane];
    out[base + lane + 32] = (v1 - m) * inv * g[lane + 32] + bt[lane + 32];
}

// in-place log_softmax over last dim 64, wave per row
__global__ __launch_bounds__(256) void logsoftmax_kernel(float* __restrict__ x, int nrows)
{
    int row = blockIdx.x * 8 + (threadIdx.x >> 5);
    if (row >= nrows) return;
    int lane = threadIdx.x & 31;
    long long base = (long long)row * 64;
    float v0 = x[base + lane], v1 = x[base + lane + 32];
    float m = wave_max(fmaxf(v0, v1));
    float s = wave_sum(__expf(v0 - m) + __expf(v1 - m));
    float l = __logf(s);
    x[base + lane]      = v0 - m - l;
    x[base + lane + 32] = v1 - m - l;
}

// y[row] = dot64(out2[row], w) + b
__global__ __launch_bounds__(256) void conv2_kernel(
    const float* __restrict__ x, const float* __restrict__ w,
    const float* __restrict__ b, float* __restrict__ y, int nrows)
{
    int row = blockIdx.x * 8 + (threadIdx.x >> 5);
    if (row >= nrows) return;
    int lane = threadIdx.x & 31;
    long long base = (long long)row * 64;
    float s = wave_sum(x[base + lane] * w[lane] + x[base + lane + 32] * w[lane + 32]);
    if (lane == 0) y[row] = s + b[0];
}

// =============================== driver ====================================
extern "C" void kernel_launch(void* const* d_in, const int* in_sizes, int n_in,
                              void* d_out, int out_size, void* d_ws, size_t ws_size,
                              hipStream_t stream)
{
    const float* x       = (const float*)d_in[0];
    const float* conv1_w = (const float*)d_in[1];
    const float* conv1_b = (const float*)d_in[2];
    const float* conv2_w = (const float*)d_in[3];
    const float* conv2_b = (const float*)d_in[4];
    const float* emb1    = (const float*)d_in[5];
    const float* emb2    = (const float*)d_in[6];
    const float* glin1_w = (const float*)d_in[7];
    const float* glin1_b = (const float*)d_in[8];
    const float* glin2_w = (const float*)d_in[9];
    const float* glin2_b = (const float*)d_in[10];
    const float* embl_w  = (const float*)d_in[11];
    const float* embl_b  = (const float*)d_in[12];
    const float* sq_w    = (const float*)d_in[13];
    const float* sk_w    = (const float*)d_in[14];
    const float* sv_w    = (const float*)d_in[15];
    const float* sfc_w   = (const float*)d_in[16];
    const float* sfc_b   = (const float*)d_in[17];
    const float* sn1_g   = (const float*)d_in[18];
    const float* sn1_b   = (const float*)d_in[19];
    const float* sn2_g   = (const float*)d_in[20];
    const float* sn2_b   = (const float*)d_in[21];
    const float* tn1_g   = (const float*)d_in[22];
    const float* tn1_b   = (const float*)d_in[23];
    const float* tn2_g   = (const float*)d_in[24];
    const float* tn2_b   = (const float*)d_in[25];
    const float* bn1_g   = (const float*)d_in[26];
    const float* bn1_b   = (const float*)d_in[27];
    const float* bn2_g   = (const float*)d_in[28];
    const float* bn2_b   = (const float*)d_in[29];
    const float* sff_w1  = (const float*)d_in[30];
    const float* sff_b1  = (const float*)d_in[31];
    const float* sff_w2  = (const float*)d_in[32];
    const float* sff_b2  = (const float*)d_in[33];
    const float* gc1_w   = (const float*)d_in[34];
    const float* gc1_b   = (const float*)d_in[35];
    const float* gc2_w   = (const float*)d_in[36];
    const float* gc2_b   = (const float*)d_in[37];
    const float* fs_w    = (const float*)d_in[38];
    const float* fs_b    = (const float*)d_in[39];
    const float* fg_w    = (const float*)d_in[40];
    const float* fg_b    = (const float*)d_in[41];
    const float* temb    = (const float*)d_in[42];
    const float* tq_w    = (const float*)d_in[43];
    const float* tq_b    = (const float*)d_in[44];
    const float* tk_w    = (const float*)d_in[45];
    const float* tk_b    = (const float*)d_in[46];
    const float* tv_w    = (const float*)d_in[47];
    const float* tv_b    = (const float*)d_in[48];
    const float* tfc_w   = (const float*)d_in[49];
    const float* tfc_b   = (const float*)d_in[50];
    const float* tff_w1  = (const float*)d_in[51];
    const float* tff_b1  = (const float*)d_in[52];
    const float* tff_w2  = (const float*)d_in[53];
    const float* tff_b2  = (const float*)d_in[54];
    float* y = (float*)d_out;

    // ---- scratch arena (floats); slots reused across phases ----
    const long long S1 = (long long)BNT * NE;   // 7.68M floats
    float* ws   = (float*)d_ws;
    float* q    = ws;                 // q  -> later xt
    float* xsb  = ws + S1;            // xs -> f2 -> t_attn
    float* R1   = ws + 2 * S1;        // 4*S1: t1/h1 -> Qh,Kh,Vh,O -> f1 -> g1,g2 -> Qt,Kt,Vt,Ot -> f1t
    float* slotE = ws + 6 * S1;       // t2 -> U_t
    float* slotF = ws + 7 * S1;       // XG -> M_t
    float* slotG = ws + 8 * S1;       // attn_out -> U_s -> f2t -> out2
    float* slotH = ws + 9 * S1;       // M_s -> out1
    float* sm    = ws + 10 * S1;      // small region
    float* nv1  = sm;                 sm += NN * NE;
    float* nv2  = sm;                 sm += NN * NE;
    float* M1   = sm;                 sm += NN * NN;
    float* amat = sm;                 sm += NN * NN;
    float* adj  = sm;                 sm += NN * NN;
    float* adjn = sm;                 sm += NN * NN;
    float* work = sm;                 sm += NN * NN;
    float* DS   = sm;                 sm += NN * NE;
    float* st   = sm;                 sm += 2;

    const int rows = BNT;
    const dim3 b256(256);
    #define ROWBLK(n) dim3(((n) + 7) / 8)
    #define ELBLK(n)  dim3(((n) + 255) / 256)

    // 1) conv1 -> q [B,N,T,E]
    conv1_kernel<<<ELBLK(BNT * NE), b256, 0, stream>>>(x, conv1_w, conv1_b, q, BNT * NE);

    // 2) graph constructor
    gemm(stream, emb1, glin1_w, nv1, glin1_b, NN, NE, NE, 64, 64, 64, 1, 0, 0, 0, 0, 2, 0);
    gemm(stream, emb2, glin2_w, nv2, glin2_b, NN, NE, NE, 64, 64, 64, 1, 0, 0, 0, 0, 2, 0);
    gemm(stream, nv1, nv2, M1, nullptr, NN, NN, NE, 64, 64, NN, 1, 0, 0, 0, 1, 0, 0); // nv1@nv2^T
    a_build_kernel<<<ELBLK(NN * NN), b256, 0, stream>>>(M1, amat);
    topk_kernel<<<1, 192, 0, stream>>>(amat, adj, work);
    gemm(stream, adj, embl_w, DS, embl_b, NN, NE, NN, NN, 64, 64, 1, 0, 0, 0, 0, 0, 0);
    stats_kernel<<<1, 256, 0, stream>>>(adj, st);
    adjnorm_kernel<<<ELBLK(NN * NN), b256, 0, stream>>>(adj, st, adjn);

    // 3) GCN:  h1 = relu(adjn @ (q@gc1_w) + b1) ; h2 = adjn @ (h1@gc2_w) + b2
    float* t1 = R1;            // [B,N,T,128]
    float* h1 = R1 + 2 * S1;   // [B,N,T,128]
    gemm(stream, q, gc1_w, t1, nullptr, rows, 128, 64, 64, 128, 128, 1, 0, 0, 0, 0, 0, 0);
    gemm(stream, adjn, t1, h1, gc1_b, NN, NT * 128, NN, NN, NT * 128, NT * 128,
         NB, 0, (long long)NN * NT * 128, (long long)NN * NT * 128, 0, 1, 128);
    float* t2 = slotE;
    gemm(stream, h1, gc2_w, t2, nullptr, rows, 64, 128, 128, 64, 64, 1, 0, 0, 0, 0, 0, 0);
    float* XG = slotF;
    gemm(stream, adjn, t2, XG, gc2_b, NN, NT * 64, NN, NN, NT * 64, NT * 64,
         NB, 0, (long long)NN * NT * 64, (long long)NN * NT * 64, 0, 0, 64);
    logsoftmax_kernel<<<ROWBLK(rows), b256, 0, stream>>>(XG, rows);

    // 4) spatial attention
    add_bcast_kernel<<<ELBLK(BNT * NE), b256, 0, stream>>>(q, DS, xsb, BNT * NE, 0);
    float* Qh = R1, *Kh = R1 + S1, *Vh = R1 + 2 * S1, *Oo = R1 + 3 * S1;
    gemm(stream, xsb, sq_w, Qh, nullptr, BNTH, NPD, NPD, 32, 32, 32, 1, 0, 0, 0, 0, 0, 0);
    gemm(stream, xsb, sk_w, Kh, nullptr, BNTH, NPD, NPD, 32, 32, 32, 1, 0, 0, 0, 0, 0, 0);
    gemm(stream, xsb, sv_w, Vh, nullptr, BNTH, NPD, NPD, 32, 32, 32, 1, 0, 0, 0, 0, 0, 0);
    attn_fused_kernel<160><<<NB * NT * NH, b256, 0, stream>>>(
        Qh, Kh, Vh, Oo, NN, NT * NH, NH,
        (long long)NN * NT * NH * NPD, (long long)NH * NPD, (long long)NPD,
        NT * NH * NPD, 0.125f);
    float* attn_out = slotG;
    gemm(stream, Oo, sfc_w, attn_out, sfc_b, rows, 64, 64, 64, 64, 64, 1, 0, 0, 0, 0, 0, 0);
    float* Ms = slotH;
    ln_kernel<<<ROWBLK(rows), b256, 0, stream>>>(attn_out, xsb, nullptr, nullptr,
                                                 sn1_g, sn1_b, Ms, rows);
    float* f1 = R1;   // [rows,256] = 4*S1
    gemm(stream, Ms, sff_w1, f1, sff_b1, rows, 256, 64, 64, 256, 256, 1, 0, 0, 0, 0, 1, 0);
    float* f2 = xsb;  // xs dead
    gemm(stream, f1, sff_w2, f2, sff_b2, rows, 64, 256, 256, 64, 64, 1, 0, 0, 0, 0, 0, 0);
    float* Us = slotG; // attn_out dead
    ln_kernel<<<ROWBLK(rows), b256, 0, stream>>>(f2, Ms, nullptr, nullptr,
                                                 sn2_g, sn2_b, Us, rows);
    float* g1 = R1, *g2 = R1 + S1;  // f1 dead
    gemm(stream, Us, fs_w, g1, fs_b, rows, 64, 64, 64, 64, 64, 1, 0, 0, 0, 0, 0, 0);
    gemm(stream, XG, fg_w, g2, fg_b, rows, 64, 64, 64, 64, 64, 1, 0, 0, 0, 0, 0, 0);
    float* out1 = slotH; // Ms dead
    gate_ln_kernel<<<ROWBLK(rows), b256, 0, stream>>>(g1, g2, Us, XG, q,
                                                      bn1_g, bn1_b, out1, rows);

    // 5) temporal attention
    float* xt = q;  // q dead
    add_bcast_kernel<<<ELBLK(BNT * NE), b256, 0, stream>>>(out1, temb, xt, BNT * NE, 1);
    float* Qt = R1, *Kt = R1 + S1, *Vt = R1 + 2 * S1, *Ot = R1 + 3 * S1;
    gemm(stream, xt, tq_w, Qt, tq_b, BNTH, NPD, NPD, 32, 32, 32, 1, 0, 0, 0, 0, 0, 0);
    gemm(stream, xt, tk_w, Kt, tk_b, BNTH, NPD, NPD, 32, 32, 32, 1, 0, 0, 0, 0, 0, 0);
    gemm(stream, xt, tv_w, Vt, tv_b, BNTH, NPD, NPD, 32, 32, 32, 1, 0, 0, 0, 0, 0, 0);
    attn_fused_kernel<112><<<NB * NN * NH, b256, 0, stream>>>(
        Qt, Kt, Vt, Ot, NT, NN * NH, NH,
        (long long)NN * NT * NH * NPD, (long long)NT * NH * NPD, (long long)NPD,
        NH * NPD, 0.125f);
    float* t_attn = xsb;  // f2 dead
    gemm(stream, Ot, tfc_w, t_attn, tfc_b, rows, 64, 64, 64, 64, 64, 1, 0, 0, 0, 0, 0, 0);
    float* Mt = slotF;  // XG dead
    ln_kernel<<<ROWBLK(rows), b256, 0, stream>>>(t_attn, xt, nullptr, nullptr,
                                                 tn1_g, tn1_b, Mt, rows);
    float* f1t = R1;  // Qt..Ot dead
    gemm(stream, Mt, tff_w1, f1t, tff_b1, rows, 256, 64, 64, 256, 256, 1, 0, 0, 0, 0, 1, 0);
    float* f2t = slotG;  // Us dead
    gemm(stream, f1t, tff_w2, f2t, tff_b2, rows, 64, 256, 256, 64, 64, 1, 0, 0, 0, 0, 0, 0);
    float* Ut = slotE;  // t2 dead
    ln_kernel<<<ROWBLK(rows), b256, 0, stream>>>(Mt, f2t, nullptr, nullptr,
                                                 tn2_g, tn2_b, Ut, rows);
    float* out2 = slotG;  // f2t consumed
    ln_kernel<<<ROWBLK(rows), b256, 0, stream>>>(Ut, xt, Mt, out1,
                                                 bn2_g, bn2_b, out2, rows);

    // 6) conv2 -> y [B,N,T]
    conv2_kernel<<<ROWBLK(rows), b256, 0, stream>>>(out2, conv2_w, conv2_b, y, rows);

    (void)in_sizes; (void)n_in; (void)out_size; (void)ws_size;
}
